// CompressedViT_12275016532547
// MI455X (gfx1250) — compile-verified, matrix operand order checked by reading, not run on hardware
//
#include <hip/hip_runtime.h>
#include <hip/hip_bf16.h>
#include <cmath>

typedef _Float16 f16;
typedef f16  f16x16 __attribute__((ext_vector_type(16)));
typedef f16  f16x8  __attribute__((ext_vector_type(8)));
typedef float f32x8 __attribute__((ext_vector_type(8)));
typedef int  i32x4  __attribute__((__vector_size__(16)));

#define BMT 128
#define BNT 128
#define BKT 32
#define LDSW 40   // padded LDS row stride (f16): 80B/row; b128 chunks stay 16B aligned

#if defined(__has_builtin)
# if __has_builtin(__builtin_amdgcn_global_load_async_to_lds_b128)
#  define HAVE_ASYNC_LDS 1
# endif
#endif

__device__ inline void async_wait0()
{
#if defined(__has_builtin)
# if __has_builtin(__builtin_amdgcn_s_wait_asynccnt)
    __builtin_amdgcn_s_wait_asynccnt(0);
# else
    asm volatile("s_wait_asynccnt 0x0" ::: "memory");
# endif
#else
    asm volatile("s_wait_asynccnt 0x0" ::: "memory");
#endif
}

__device__ inline void async_wait4()
{
#if defined(__has_builtin)
# if __has_builtin(__builtin_amdgcn_s_wait_asynccnt)
    __builtin_amdgcn_s_wait_asynccnt(4);
# else
    asm volatile("s_wait_asynccnt 0x4" ::: "memory");
# endif
#else
    asm volatile("s_wait_asynccnt 0x4" ::: "memory");
#endif
}

#ifdef HAVE_ASYNC_LDS
__device__ inline void async_copy16(const f16* gsrc, f16* ldst)
{
    __builtin_amdgcn_global_load_async_to_lds_b128(
        (__attribute__((address_space(1))) i32x4*)const_cast<f16*>(gsrc),
        (__attribute__((address_space(3))) i32x4*)ldst,
        0, 0);
}
#endif

// ---------------------------------------------------------------------------
// Strided-batched GEMM:  C[z] = act(A[z] (MxK) * B[z]^T (NcxK) + bias (+R))
// A,B are f16 in memory; accumulate fp32; C stored fp32 (Cf) or f16 (Ch).
// A(m,k) = A[zb*sAb + zh*sAh + m*sAm + k]          (A is always K-contiguous)
// B(n,k) = B[zb*sBb + zh*sBh + n*sBn + k*sBk]      (BVEC => sBk==1 & aligned)
// C(m,n) = C[zb*sCb + zh*sCh + m*sCm + n]
// Fast path (interior tiles, K%32==0): double-buffered async global->LDS
// staging overlapped with WMMA; waits asynccnt<=4 so next tile stays in
// flight behind compute. Edge tiles / strided B use guarded element path.
// Block 256 thr = 8 waves; tile 128x128x32; wave tile 64x32 (4x2 wmma frags).
// ---------------------------------------------------------------------------
template<bool AVEC, bool BVEC>
__global__ __launch_bounds__(256)
void vit_gemm_f16(const f16* __restrict__ A, const f16* __restrict__ Bw,
                  const float* __restrict__ bias, const float* __restrict__ R,
                  float* __restrict__ Cf, f16* __restrict__ Ch,
                  int M, int Nc, int K,
                  long long sAm, long long sAb, long long sAh,
                  long long sBn, long long sBk, long long sBb, long long sBh,
                  long long sCm, long long sCb, long long sCh,
                  int nh, int act)
{
    __shared__ __align__(16) f16 As[2 * BMT * LDSW];
    __shared__ __align__(16) f16 Bs[2 * BNT * LDSW];

    const int tid  = threadIdx.x;
    const int lane = tid & 31;
    const int wave = tid >> 5;
    const int zb = blockIdx.z / nh;
    const int zh = blockIdx.z % nh;
    const long long aB = (long long)zb * sAb + (long long)zh * sAh;
    const long long bB = (long long)zb * sBb + (long long)zh * sBh;
    const long long cB = (long long)zb * sCb + (long long)zh * sCh;
    const int m0 = blockIdx.x * BMT;
    const int n0 = blockIdx.y * BNT;
    const int wm = (wave >> 2) * 64;   // wave grid 2(m) x 4(n)
    const int wn = (wave & 3) * 32;

    // staging: 4 threads per row, 8 consecutive k (16B) each, 2 row passes
    const int sk8 = (tid & 3) * 8;     // 0,8,16,24
    const int sr  = tid >> 2;          // 0..63

    const f16* rowA[2]; bool okA[2];
    const f16* rowB[2]; bool okB[2];
    #pragma unroll
    for (int p = 0; p < 2; ++p) {
        int gm = m0 + sr + p * 64;
        okA[p] = gm < M;
        rowA[p] = A + aB + (long long)(okA[p] ? gm : (M - 1)) * sAm;
        int gn = n0 + sr + p * 64;
        okB[p] = gn < Nc;
        rowB[p] = Bw + bB + (long long)(okB[p] ? gn : (Nc - 1)) * sBn;
    }
    const bool cleanA = (m0 + BMT <= M);
    const bool cleanB = (n0 + BNT <= Nc);

    f32x8 acc[4][2] = {};

    // ---- fragment compute from LDS buffer `buf`
    // 16-bit 16x32 layout: lane%16 = row, lane/16 = K-half;
    // two b128 chunks per fragment at +0B and +32B within the LDS row.
    auto compute = [&](int buf) {
        const int fr = lane & 15, hf = lane >> 4;
        const f16* Ab = As + buf * (BMT * LDSW);
        const f16* Bb = Bs + buf * (BNT * LDSW);
        f16x16 af[4], bf[2];
        #pragma unroll
        for (int i = 0; i < 4; ++i) {
            const f16* p = &Ab[(wm + i * 16 + fr) * LDSW + hf * 8];
            ((f16x8*)&af[i])[0] = *(const f16x8*)p;
            ((f16x8*)&af[i])[1] = *(const f16x8*)(p + 16);
        }
        #pragma unroll
        for (int j = 0; j < 2; ++j) {
            const f16* p = &Bb[(wn + j * 16 + fr) * LDSW + hf * 8];
            ((f16x8*)&bf[j])[0] = *(const f16x8*)p;
            ((f16x8*)&bf[j])[1] = *(const f16x8*)(p + 16);
        }
        #pragma unroll
        for (int i = 0; i < 4; ++i)
            #pragma unroll
            for (int j = 0; j < 2; ++j)
                acc[i][j] = __builtin_amdgcn_wmma_f32_16x16x32_f16(
                    false, af[i], false, bf[j], (short)0, acc[i][j], false, false);
    };

    if (AVEC && BVEC && cleanA && cleanB && ((K & (BKT - 1)) == 0)) {
        // ================= fast path: double-buffered async pipeline ========
        auto stage = [&](int k0, int buf) {
            #pragma unroll
            for (int p = 0; p < 2; ++p) {
#ifdef HAVE_ASYNC_LDS
                async_copy16(rowA[p] + k0 + sk8,
                             &As[buf * (BMT * LDSW) + (sr + p * 64) * LDSW + sk8]);
#else
                *(f16x8*)&As[buf * (BMT * LDSW) + (sr + p * 64) * LDSW + sk8] =
                    *(const f16x8*)(rowA[p] + k0 + sk8);
#endif
            }
            #pragma unroll
            for (int p = 0; p < 2; ++p) {
#ifdef HAVE_ASYNC_LDS
                async_copy16(rowB[p] + k0 + sk8,
                             &Bs[buf * (BNT * LDSW) + (sr + p * 64) * LDSW + sk8]);
#else
                *(f16x8*)&Bs[buf * (BNT * LDSW) + (sr + p * 64) * LDSW + sk8] =
                    *(const f16x8*)(rowB[p] + k0 + sk8);
#endif
            }
        };

        stage(0, 0);
        int buf = 0;
        for (int k0 = 0; k0 < K; k0 += BKT, buf ^= 1) {
            if (k0 + BKT < K) {
                stage(k0 + BKT, buf ^ 1);   // next tile fills other buffer
#ifdef HAVE_ASYNC_LDS
                async_wait4();              // current tile resident; next in flight
#endif
            } else {
#ifdef HAVE_ASYNC_LDS
                async_wait0();
#endif
            }
            __syncthreads();
            compute(buf);
            __syncthreads();
        }
    } else {
        // ================= general path: guarded element staging ============
        for (int k0 = 0; k0 < K; k0 += BKT) {
            #pragma unroll
            for (int p = 0; p < 2; ++p) {
                f16x8 hv;
                #pragma unroll
                for (int e = 0; e < 8; ++e) {
                    int gk = k0 + sk8 + e;
                    int kc = gk < K ? gk : (K - 1);
                    f16 v = rowA[p][kc];
                    if (!(okA[p] && gk < K)) v = (f16)0.0f;
                    hv[e] = v;
                }
                *(f16x8*)&As[(sr + p * 64) * LDSW + sk8] = hv;
            }
            #pragma unroll
            for (int p = 0; p < 2; ++p) {
                f16x8 hv;
                #pragma unroll
                for (int e = 0; e < 8; ++e) {
                    int gk = k0 + sk8 + e;
                    int kc = gk < K ? gk : (K - 1);
                    long long off = BVEC ? (long long)kc : (long long)kc * sBk;
                    f16 v = rowB[p][off];
                    if (!(okB[p] && gk < K)) v = (f16)0.0f;
                    hv[e] = v;
                }
                *(f16x8*)&Bs[(sr + p * 64) * LDSW + sk8] = hv;
            }
            __syncthreads();
            compute(0);
            __syncthreads();
        }
    }

    // ---- epilogue: C/D layout: VGPR g, lane l -> m = 8*(l/16)+g, n = l%16
    const int lrow = 8 * (lane >> 4);
    const int lcol = lane & 15;
    #pragma unroll
    for (int i = 0; i < 4; ++i) {
        #pragma unroll
        for (int j = 0; j < 2; ++j) {
            int nn = n0 + wn + j * 16 + lcol;
            if (nn >= Nc) continue;
            float bv = bias ? bias[nn] : 0.0f;
            #pragma unroll
            for (int g = 0; g < 8; ++g) {
                int mm = m0 + wm + i * 16 + lrow + g;
                if (mm >= M) continue;
                float v = acc[i][j][g] + bv;
                long long off = cB + (long long)mm * sCm + nn;
                if (R) v += R[off];
                if (act == 1) v = 0.5f * v * (1.0f + erff(v * 0.70710678118654752f));
                if (Ch) Ch[off] = (f16)v;
                else    Cf[off] = v;
            }
        }
    }
}

// --------------------------------------------------------------------------
__global__ __launch_bounds__(256)
void vit_f32_to_f16(const float* __restrict__ in, f16* __restrict__ outp,
                    long long n)
{
    long long i = (long long)blockIdx.x * 256 + threadIdx.x;
    if (i < n) outp[i] = (f16)in[i];
}

// --------------------------------------------------------------------------
// im2row: x[32,3,224,224] -> xp[32*196, 768] (f16), col = c*256 + py*16 + px
// --------------------------------------------------------------------------
__global__ __launch_bounds__(256)
void vit_im2row(const float* __restrict__ x, f16* __restrict__ xp)
{
    long long idx = (long long)blockIdx.x * 256 + threadIdx.x;
    if (idx >= (long long)6272 * 768) return;
    int kcol = (int)(idx % 768);
    long long rowi = idx / 768;
    int p = (int)(rowi % 196);
    int b = (int)(rowi / 196);
    int c  = kcol >> 8;
    int py = (kcol >> 4) & 15;
    int px = kcol & 15;
    int gy = p / 14, gx = p % 14;
    xp[idx] = (f16)x[((long long)(b * 3 + c) * 224 + gy * 16 + py) * 224 + gx * 16 + px];
}

// --------------------------------------------------------------------------
// cls token + positional embedding on fp32 residual stream h[32,197,768]
// --------------------------------------------------------------------------
__global__ __launch_bounds__(256)
void vit_cls_pos(float* __restrict__ h, const float* __restrict__ cls,
                 const float* __restrict__ pos)
{
    long long idx = (long long)blockIdx.x * 256 + threadIdx.x;
    if (idx >= (long long)6304 * 768) return;
    int e = (int)(idx % 768);
    long long row = idx / 768;
    int t = (int)(row % 197);
    if (t == 0) h[idx] = cls[e] + pos[e];
    else        h[idx] += pos[(long long)t * 768 + e];
}

// --------------------------------------------------------------------------
// LayerNorm over E=768 (fp32 in, f16 out); one 256-thread block per row
// --------------------------------------------------------------------------
__global__ __launch_bounds__(256)
void vit_layernorm768(const float* __restrict__ x, const float* __restrict__ w,
                      const float* __restrict__ b, f16* __restrict__ y)
{
    const long long row = blockIdx.x;
    const float* xr = x + row * 768;
    f16* yr = y + row * 768;
    const int t = threadIdx.x;
    float v0 = xr[t], v1 = xr[t + 256], v2 = xr[t + 512];
    float s = v0 + v1 + v2;
    __shared__ float r1[8], r2[8];
    #pragma unroll
    for (int o = 16; o > 0; o >>= 1) s += __shfl_xor(s, o, 32);
    if ((t & 31) == 0) r1[t >> 5] = s;
    __syncthreads();
    float mu = (r1[0] + r1[1] + r1[2] + r1[3] + r1[4] + r1[5] + r1[6] + r1[7])
               * (1.0f / 768.0f);
    float d0 = v0 - mu, d1 = v1 - mu, d2 = v2 - mu;
    float q = d0 * d0 + d1 * d1 + d2 * d2;
    #pragma unroll
    for (int o = 16; o > 0; o >>= 1) q += __shfl_xor(q, o, 32);
    if ((t & 31) == 0) r2[t >> 5] = q;
    __syncthreads();
    float var = (r2[0] + r2[1] + r2[2] + r2[3] + r2[4] + r2[5] + r2[6] + r2[7])
                * (1.0f / 768.0f);
    float inv = rsqrtf(var + 1e-6f);
    yr[t]       = (f16)(d0 * inv * w[t]       + b[t]);
    yr[t + 256] = (f16)(d1 * inv * w[t + 256] + b[t + 256]);
    yr[t + 512] = (f16)(d2 * inv * w[t + 512] + b[t + 512]);
}

// --------------------------------------------------------------------------
// Row softmax over 197 cols (fp32 logits -> f16 probs), pre-scale folded in.
// --------------------------------------------------------------------------
__global__ __launch_bounds__(256)
void vit_softmax197(const float* __restrict__ S, f16* __restrict__ P,
                    int nrows, float scale)
{
    const int wave = threadIdx.x >> 5;
    const int lane = threadIdx.x & 31;
    const int row = blockIdx.x * 8 + wave;
    if (row >= nrows) return;
    const float* r = S + (long long)row * 197;
    f16* po = P + (long long)row * 197;
    float vals[7];
    float mx = -3.4e38f;
    #pragma unroll
    for (int i = 0; i < 7; ++i) {
        int c = lane + i * 32;
        float v = (c < 197) ? r[c] * scale : -3.4e38f;
        vals[i] = v;
        mx = fmaxf(mx, v);
    }
    #pragma unroll
    for (int o = 16; o > 0; o >>= 1) mx = fmaxf(mx, __shfl_xor(mx, o, 32));
    float sum = 0.0f;
    #pragma unroll
    for (int i = 0; i < 7; ++i) {
        int c = lane + i * 32;
        float e = (c < 197) ? expf(vals[i] - mx) : 0.0f;
        vals[i] = e;
        sum += e;
    }
    #pragma unroll
    for (int o = 16; o > 0; o >>= 1) sum += __shfl_xor(sum, o, 32);
    float inv = 1.0f / sum;
    #pragma unroll
    for (int i = 0; i < 7; ++i) {
        int c = lane + i * 32;
        if (c < 197) po[c] = (f16)(vals[i] * inv);
    }
}

// --------------------------------------------------------------------------
extern "C" void kernel_launch(void* const* d_in, const int* in_sizes, int n_in,
                              void* d_out, int out_size, void* d_ws, size_t ws_size,
                              hipStream_t stream)
{
    (void)in_sizes; (void)n_in; (void)out_size; (void)ws_size;
    const float* x       = (const float*)d_in[0];
    const float* patch_w = (const float*)d_in[1];
    const float* patch_b = (const float*)d_in[2];
    const float* cls_tok = (const float*)d_in[3];
    const float* pos     = (const float*)d_in[4];
    const float* qkv_w   = (const float*)d_in[5];
    const float* qkv_b   = (const float*)d_in[6];
    const float* proj_w  = (const float*)d_in[7];
    const float* proj_b  = (const float*)d_in[8];
    const float* ln1_w   = (const float*)d_in[9];
    const float* ln1_b   = (const float*)d_in[10];
    const float* ln2_w   = (const float*)d_in[11];
    const float* ln2_b   = (const float*)d_in[12];
    const float* fc1_wp  = (const float*)d_in[13];
    const float* fc1_bp  = (const float*)d_in[14];
    const float* fc2_wp  = (const float*)d_in[15];
    const float* fc2_bp  = (const float*)d_in[16];
    const float* norm_w  = (const float*)d_in[17];
    const float* norm_b  = (const float*)d_in[18];
    const float* head_w  = (const float*)d_in[19];
    const float* head_b  = (const float*)d_in[20];
    float* out = (float*)d_out;

    const int B = 32, E = 768, NH = 12, HD = 64, DFF = 3072, L = 12;
    const int N = 197, NP = 196, ROWS = B * N;   // 6304
    const float SCALE = 0.125f;                  // 64^-0.5

    // workspace layout: fp32 first, then f16 regions (all 16B aligned)
    char* base = (char*)d_ws;
    float* h  = (float*)base;   base += (size_t)ROWS * E * 4;
    float* S  = (float*)base;   base += (size_t)B * NH * N * N * 4;
    f16* xph  = (f16*)base;     base += (size_t)B * NP * E * 2;
    f16* yh   = (f16*)base;     base += (size_t)ROWS * E * 2;
    f16* qkvh = (f16*)base;     base += (size_t)ROWS * 3 * E * 2;
    f16* Ph   = (f16*)base;     base += (size_t)B * NH * N * N * 2;
    f16* obh  = (f16*)base;     base += (size_t)ROWS * E * 2;
    f16* mlph = (f16*)base;     base += (size_t)ROWS * DFF * 2;
    f16* wq   = (f16*)base;     base += (size_t)3 * E * E * 2;   // also patch/head scratch
    f16* wp   = (f16*)base;     base += (size_t)E * E * 2;
    f16* w1h  = (f16*)base;     base += (size_t)DFF * E * 2;
    f16* w2h  = (f16*)base;

    auto cvt = [&](const float* src, f16* dst, long long n) {
        vit_f32_to_f16<<<(int)((n + 255) / 256), 256, 0, stream>>>(src, dst, n);
    };
    auto gemm = [&](const f16* A, long long sAm, long long sAb, long long sAh,
                    const f16* Bw, long long sBn, long long sBk, long long sBb, long long sBh,
                    const float* bias, const float* R,
                    float* Cf, f16* Ch, long long sCm, long long sCb, long long sCh,
                    int M, int Nc, int K, int batB, int nhv, int act, bool fullvec) {
        dim3 g((M + BMT - 1) / BMT, (Nc + BNT - 1) / BNT, batB * nhv);
        if (fullvec)
            vit_gemm_f16<true, true><<<g, 256, 0, stream>>>(A, Bw, bias, R, Cf, Ch,
                M, Nc, K, sAm, sAb, sAh, sBn, sBk, sBb, sBh, sCm, sCb, sCh, nhv, act);
        else
            vit_gemm_f16<false, false><<<g, 256, 0, stream>>>(A, Bw, bias, R, Cf, Ch,
                M, Nc, K, sAm, sAb, sAh, sBn, sBk, sBb, sBh, sCm, sCb, sCh, nhv, act);
    };

    // 1. patch extraction (fp32 -> f16 rows)
    vit_im2row<<<(int)(((long long)B * NP * E + 255) / 256), 256, 0, stream>>>(x, xph);

    // 2. patch embed: tok = xp @ patch_w.T + patch_b -> h rows t>=1 (fp32)
    cvt(patch_w, wq, (long long)E * E);
    gemm(xph, E, (long long)NP * E, 0,
         wq, E, 1, 0, 0,
         patch_b, nullptr,
         h + E, nullptr, E, (long long)N * E, 0,
         NP, E, E, B, 1, 0, true);

    // 3. cls token + pos embed
    vit_cls_pos<<<(int)(((long long)ROWS * E + 255) / 256), 256, 0, stream>>>(h, cls_tok, pos);

    for (int l = 0; l < L; ++l) {
        cvt(qkv_w  + (size_t)l * 3 * E * E, wq,  (long long)3 * E * E);
        cvt(proj_w + (size_t)l * E * E,     wp,  (long long)E * E);
        cvt(fc1_wp + (size_t)l * DFF * E,   w1h, (long long)DFF * E);
        cvt(fc2_wp + (size_t)l * E * DFF,   w2h, (long long)E * DFF);
        const float* qb = qkv_b  + (size_t)l * 3 * E;
        const float* pb = proj_b + (size_t)l * E;
        const float* b1 = fc1_bp + (size_t)l * DFF;
        const float* b2 = fc2_bp + (size_t)l * E;

        // LN1 -> f16
        vit_layernorm768<<<ROWS, 256, 0, stream>>>(h, ln1_w + (size_t)l * E,
                                                   ln1_b + (size_t)l * E, yh);
        // QKV (f16 out): [6304,768] x [2304,768]^T
        gemm(yh, E, 0, 0,  wq, E, 1, 0, 0,  qb, nullptr,
             nullptr, qkvh, 3 * E, 0, 0,  ROWS, 3 * E, E, 1, 1, 0, true);

        // scores (fp32 out): per (b,h): S = Q(197x64) @ K^T(64x197)
        gemm(qkvh,     3 * E, (long long)N * 3 * E, HD,
             qkvh + E, 3 * E, 1, (long long)N * 3 * E, HD,
             nullptr, nullptr,
             S, nullptr, N, (long long)NH * N * N, (long long)N * N,
             N, N, HD, B, NH, 0, true);

        // softmax: fp32 logits -> f16 probs (scale folded in)
        vit_softmax197<<<(B * NH * N + 7) / 8, 256, 0, stream>>>(S, Ph, B * NH * N, SCALE);

        // O (f16 out) = P(197x197) @ V(197x64); B = V^T via strides
        gemm(Ph,          N, (long long)NH * N * N, (long long)N * N,
             qkvh + 2 * E, 1, 3 * E, (long long)N * 3 * E, HD,
             nullptr, nullptr,
             nullptr, obh, E, (long long)N * E, HD,
             N, HD, N, B, NH, 0, false);

        // proj + residual (fp32): h = h + ob @ pw^T + pb
        gemm(obh, E, 0, 0,  wp, E, 1, 0, 0,  pb, h,
             h, nullptr, E, 0, 0,  ROWS, E, E, 1, 1, 0, true);

        // LN2 -> f16
        vit_layernorm768<<<ROWS, 256, 0, stream>>>(h, ln2_w + (size_t)l * E,
                                                   ln2_b + (size_t)l * E, yh);
        // FC1 + exact GELU (f16 out)
        gemm(yh, E, 0, 0,  w1h, E, 1, 0, 0,  b1, nullptr,
             nullptr, mlph, DFF, 0, 0,  ROWS, DFF, E, 1, 1, 1, true);
        // FC2 + residual (fp32)
        gemm(mlph, DFF, 0, 0,  w2h, DFF, 1, 0, 0,  b2, h,
             h, nullptr, E, 0, 0,  ROWS, E, DFF, 1, 1, 0, true);
    }

    // final norm -> f16 (only cls rows consumed)
    vit_layernorm768<<<ROWS, 256, 0, stream>>>(h, norm_w, norm_b, yh);

    // head: out[b, nc] = y[row b*197] @ head_w^T + head_b (fp32 out)
    cvt(head_w, wq, (long long)1000 * E);
    gemm(yh, (long long)N * E, 0, 0,  wq, E, 1, 0, 0,  head_b, nullptr,
         out, nullptr, 1000, 0, 0,  B, 1000, E, 1, 1, 0, true);
}
